// SimpleGATLayer_36094905155714
// MI455X (gfx1250) — matrix operand order repaired and checked by reading, not compiled, not run on hardware
//
#include <hip/hip_runtime.h>
#include <hip/hip_bf16.h>

#define NN    8192
#define INF_  512
#define OUTF  256
#define ALPHA 0.2f
#define NEGINF -9000000000000000.0f

typedef __attribute__((ext_vector_type(2)))  float  v2f;
typedef __attribute__((ext_vector_type(8)))  float  v8f;
typedef __attribute__((ext_vector_type(16))) __bf16 v16bf;

union BfVecU { unsigned int u[8]; v16bf v; };
union BfVecQ { uint4 q[2];        v16bf v; };

__device__ __forceinline__ float lrelu(float x) { return x > 0.0f ? x : ALPHA * x; }

__device__ __forceinline__ unsigned short f2bf(float f) {
  union { float f; unsigned int u; } c; c.f = f;
  unsigned int r = c.u + 0x7FFFu + ((c.u >> 16) & 1u);   // round-nearest-even
  return (unsigned short)(r >> 16);
}

// Async copy 8 bytes global -> LDS (CDNA5 GLOBAL_LOAD_ASYNC_TO_LDS_B64).
__device__ __forceinline__ void async_g2l_b64(void* lds_ptr, const void* gptr) {
  unsigned int lds_off = (unsigned int)(size_t)lds_ptr;      // low 32 bits = LDS offset
  unsigned long long ga = (unsigned long long)(size_t)gptr;
  asm volatile("global_load_async_to_lds_b64 %0, %1, off"
               :: "v"(lds_off), "v"(ga) : "memory");
}

// ---------------------------------------------------------------------------
// Kernel 1: Wh = h @ W in full f32 via V_WMMA_F32_16X16X4_F32.
// One wave per 16x16 output tile. Emits f32 Wh (row-major) plus a bf16
// TRANSPOSED copy WhT[n][k] so kernel 4's WMMA B operands are contiguous.
// ---------------------------------------------------------------------------
__global__ __launch_bounds__(256) void k_gemm1(const float* __restrict__ h,
                                               const float* __restrict__ W,
                                               float* __restrict__ Whf,
                                               unsigned short* __restrict__ WhT) {
  int wid  = blockIdx.x * 8 + (threadIdx.x >> 5);   // 8192 waves total
  int lane = threadIdx.x & 31;
  int mt = wid >> 4, nt = wid & 15;
  int m0 = mt * 16, n0 = nt * 16;
  int ln = lane & 15;
  int kA = (lane < 16) ? 0 : 2;

  v8f acc = {};
  const float* hrow = h + (size_t)(m0 + ln) * INF_;
  for (int k0 = 0; k0 < INF_; k0 += 4) {
    v2f av, bv;
    av.x = hrow[k0 + kA];
    av.y = hrow[k0 + kA + 1];
    const float* wp = W + (size_t)(k0 + kA) * OUTF + n0 + ln;
    bv.x = wp[0];
    bv.y = wp[OUTF];
    acc = __builtin_amdgcn_wmma_f32_16x16x4_f32(false, av, false, bv,
                                                (short)0, acc, false, false);
  }
  int mBase = m0 + ((lane < 16) ? 0 : 8);
  int nOut  = n0 + ln;
  for (int c = 0; c < 8; ++c) {
    float v = acc[c];
    Whf[(size_t)(mBase + c) * OUTF + nOut] = v;
    WhT[(size_t)nOut * NN + (mBase + c)]   = f2bf(v);   // transposed bf16
  }
}

// ---------------------------------------------------------------------------
// Kernel 2: s1 = Wh@a1, s2 = Wh@a2. One wave per row.
// ---------------------------------------------------------------------------
__global__ __launch_bounds__(256) void k_svec(const float* __restrict__ Whf,
                                              const float* __restrict__ a,
                                              float* __restrict__ s1,
                                              float* __restrict__ s2) {
  int row  = blockIdx.x * 8 + (threadIdx.x >> 5);
  int lane = threadIdx.x & 31;
  const float* wr = Whf + (size_t)row * OUTF;
  float p1 = 0.0f, p2 = 0.0f;
  for (int k = lane; k < OUTF; k += 32) {
    float v = wr[k];
    p1 += v * a[k];
    p2 += v * a[OUTF + k];
  }
  for (int o = 16; o; o >>= 1) {
    p1 += __shfl_down(p1, o, 32);
    p2 += __shfl_down(p2, o, 32);
  }
  if (lane == 0) { s1[row] = p1; s2[row] = p2; }
}

// ---------------------------------------------------------------------------
// Kernel 3: per-row softmax stats. lrelu is monotone, so
// rowmax = lrelu(s2[i] + max_{adj} s1[j]); second (L2-hot) sweep sums exp.
// Empty rows -> m=NEG_INF sentinel, Z=N (uniform attention, matching jax).
// ---------------------------------------------------------------------------
__global__ __launch_bounds__(256) void k_stats(const int* __restrict__ adj,
                                               const float* __restrict__ s1,
                                               const float* __restrict__ s2,
                                               float* __restrict__ mrow,
                                               float* __restrict__ zrow) {
  int row  = blockIdx.x * 8 + (threadIdx.x >> 5);
  int lane = threadIdx.x & 31;
  const int4*   ar  = (const int4*)(adj + (size_t)row * NN);
  const float4* s1v = (const float4*)s1;

  float mx = -3.4e38f; int cnt = 0;
  for (int it = lane; it < NN / 4; it += 32) {
    int4   av = ar[it];
    float4 sv = s1v[it];
    if (av.x > 0) { mx = fmaxf(mx, sv.x); cnt++; }
    if (av.y > 0) { mx = fmaxf(mx, sv.y); cnt++; }
    if (av.z > 0) { mx = fmaxf(mx, sv.z); cnt++; }
    if (av.w > 0) { mx = fmaxf(mx, sv.w); cnt++; }
  }
  for (int o = 16; o; o >>= 1) {
    mx   = fmaxf(mx, __shfl_xor(mx, o, 32));
    cnt += __shfl_xor(cnt, o, 32);
  }
  float s2r = s2[row];
  float m = lrelu(s2r + mx);

  float z = 0.0f;
  if (cnt > 0) {   // wave-uniform after reduction
    for (int it = lane; it < NN / 4; it += 32) {
      int4   av = ar[it];
      float4 sv = s1v[it];
      if (av.x > 0) z += __expf(lrelu(s2r + sv.x) - m);
      if (av.y > 0) z += __expf(lrelu(s2r + sv.y) - m);
      if (av.z > 0) z += __expf(lrelu(s2r + sv.z) - m);
      if (av.w > 0) z += __expf(lrelu(s2r + sv.w) - m);
    }
    for (int o = 16; o; o >>= 1) z += __shfl_xor(z, o, 32);
  }
  if (lane == 0) {
    if (cnt == 0) { mrow[row] = NEGINF; zrow[row] = (float)NN; }
    else          { mrow[row] = m;      zrow[row] = z; }
  }
}

// ---------------------------------------------------------------------------
// Kernel 4: fused  out = elu( softmax(e) @ Wh ).
// Block: 256 threads (8 waves) -> 16 rows x 256 cols. K loop over j in 32s.
//  - adj tile (2KB/step, the HBM stream) double-buffered in LDS via
//    GLOBAL_LOAD_ASYNC_TO_LDS_B64; one async op per wave per tile so
//    s_wait_asynccnt 0x1 overlaps tile t+1 fetch with tile t compute.
//  - P tile (probs, bf16) built in LDS in WMMA-A dword order (pad 17 dw).
//  - B operands read straight from pre-transposed WhT: per lane one
//    contiguous 32B run -> 2x global_load_b128, no LDS, no repack.
//  - 2x v_wmma_f32_16x16x32_bf16 per wave per step, f32 accumulate.
// ---------------------------------------------------------------------------
__global__ __launch_bounds__(256) void k_attn(const int* __restrict__ adj,
                                              const float* __restrict__ s1,
                                              const float* __restrict__ s2,
                                              const float* __restrict__ mrow,
                                              const float* __restrict__ zrow,
                                              const unsigned short* __restrict__ WhT,
                                              float* __restrict__ out) {
  __shared__ __align__(16) int ldsA[2][16 * 32];   // adj tiles, double buffered
  __shared__ unsigned int ldsP[16 * 17];           // 16x16 packed bf16 pairs (+pad)

  const int tid  = threadIdx.x;
  const int lane = tid & 31;
  const int wave = tid >> 5;
  const int ln   = lane & 15;
  const int hi   = lane >> 4;               // 0 or 1
  const int i0   = blockIdx.x * 16;

  // phase-A role: thread covers (row r, 2 adjacent cols) of the prob tile
  const int r  = tid >> 4;                  // 0..15
  const int j2 = (tid & 15) * 2;            // 0..30
  const float s2r   = s2[i0 + r];
  const float mr    = mrow[i0 + r];
  const bool  empty = (mr < -8.9e15f);

  const int* adjRow = adj + (size_t)(i0 + r) * NN + j2;

  v8f accL = {}, accH = {};
  const int n0L = wave * 32;
  const int n0H = wave * 32 + 16;
  const unsigned short* bpL = WhT + (size_t)(n0L + ln) * NN + hi * 16;
  const unsigned short* bpH = WhT + (size_t)(n0H + ln) * NN + hi * 16;

  // prologue: stage tile 0
  async_g2l_b64(&ldsA[0][r * 32 + j2], adjRow);

  for (int k0 = 0; k0 < NN; k0 += 32) {
    const int buf = (k0 >> 5) & 1;

    // issue next tile, then wait for current (<=1 outstanding keeps overlap)
    if (k0 + 32 < NN) {
      async_g2l_b64(&ldsA[buf ^ 1][r * 32 + j2], adjRow + k0 + 32);
      asm volatile("s_wait_asynccnt 0x1" ::: "memory");
    } else {
      asm volatile("s_wait_asynccnt 0x0" ::: "memory");
    }
    __syncthreads();   // adj tile visible; prior A-gathers done -> ldsP reusable

    // ---- probability tile -> LDS (bf16 pairs, WMMA-A dword order) ----
    int2  av = *(const int2*)&ldsA[buf][r * 32 + j2];
    float sa = s1[k0 + j2], sb = s1[k0 + j2 + 1];
    float p0, p1;
    if (empty) { p0 = 1.0f; p1 = 1.0f; }
    else {
      p0 = (av.x > 0) ? __expf(lrelu(s2r + sa) - mr) : 0.0f;
      p1 = (av.y > 0) ? __expf(lrelu(s2r + sb) - mr) : 0.0f;
    }
    ldsP[r * 17 + (tid & 15)] =
        (unsigned int)f2bf(p0) | ((unsigned int)f2bf(p1) << 16);
    __syncthreads();   // P tile ready

    // ---- A gather (conflict-free: 17-dword row stride) ----
    BfVecU A;
    for (int v = 0; v < 8; ++v) {
      int d = v + hi * 8;   // contiguous k = 16*hi + s, matches B
      A.u[v] = ldsP[ln * 17 + d];
    }
    // ---- B operands: contiguous 32B per lane from transposed WhT ----
    BfVecQ BL, BH;
    BL.q[0] = *(const uint4*)(bpL + k0);
    BL.q[1] = *(const uint4*)(bpL + k0 + 8);
    BH.q[0] = *(const uint4*)(bpH + k0);
    BH.q[1] = *(const uint4*)(bpH + k0 + 8);

    accL = __builtin_amdgcn_wmma_f32_16x16x32_bf16(false, A.v, false, BL.v,
                                                   (short)0, accL, false, false);
    accH = __builtin_amdgcn_wmma_f32_16x16x32_bf16(false, A.v, false, BH.v,
                                                   (short)0, accH, false, false);
  }

  // ---- epilogue: scale by 1/Z, ELU, store f32 ----
  for (int c = 0; c < 8; ++c) {
    int M = c + hi * 8;
    float invz = 1.0f / zrow[i0 + M];
    float vL = accL[c] * invz;
    float vH = accH[c] * invz;
    vL = (vL > 0.0f) ? vL : (__expf(vL) - 1.0f);
    vH = (vH > 0.0f) ? vH : (__expf(vH) - 1.0f);
    out[(size_t)(i0 + M) * OUTF + n0L + ln] = vL;
    out[(size_t)(i0 + M) * OUTF + n0H + ln] = vH;
  }
}

// ---------------------------------------------------------------------------
extern "C" void kernel_launch(void* const* d_in, const int* in_sizes, int n_in,
                              void* d_out, int out_size, void* d_ws, size_t ws_size,
                              hipStream_t stream) {
  const float* h   = (const float*)d_in[0];
  const int*   adj = (const int*)  d_in[1];
  const float* W   = (const float*)d_in[2];
  const float* a   = (const float*)d_in[3];
  float*       out = (float*)d_out;

  char* ws = (char*)d_ws;
  const size_t whElems = (size_t)NN * OUTF;           // 2,097,152
  float*          Whf  = (float*)ws;                                // 8 MB
  unsigned short* WhT  = (unsigned short*)(ws + whElems * 4);       // 4 MB (transposed bf16)
  float*          s1   = (float*)(ws + whElems * 4 + whElems * 2);  // 32 KB each
  float*          s2   = s1 + NN;
  float*          mr   = s2 + NN;
  float*          zr   = mr + NN;

  k_gemm1<<<1024, 256, 0, stream>>>(h, W, Whf, WhT);
  k_svec <<<1024, 256, 0, stream>>>(Whf, a, s1, s2);
  k_stats<<<1024, 256, 0, stream>>>(adj, s1, s2, mr, zr);
  k_attn <<< 512, 256, 0, stream>>>(adj, s1, s2, mr, zr, WhT, out);
}
